// TransformerBlock_65386582114514
// MI455X (gfx1250) — matrix operand order repaired and checked by reading, not compiled, not run on hardware
//
#include <hip/hip_runtime.h>
#include <math.h>

typedef __bf16 bf16;
typedef __attribute__((ext_vector_type(16))) __bf16 v16bf;
typedef __attribute__((ext_vector_type(8)))  __bf16 v8bf;
typedef __attribute__((ext_vector_type(8)))  float  v8f;

union AFrag { v16bf v; v8bf h[2]; };

__device__ __forceinline__ v8f vzero8() {
    v8f z; for (int i = 0; i < 8; ++i) z[i] = 0.0f; return z;
}

__device__ __forceinline__ float gelu_f(float x) {
    const float c = 0.7978845608028654f; // sqrt(2/pi)
    float x3 = x * x * x;
    return 0.5f * x * (1.0f + tanhf(c * (x + 0.044715f * x3)));
}

// ---------------------------------------------------------------- convert ---
__global__ __launch_bounds__(256) void k_f32_to_bf16(const float* __restrict__ src,
                                                     bf16* __restrict__ dst, long n) {
    long i = (long)blockIdx.x * blockDim.x + threadIdx.x;
    long stride = (long)gridDim.x * blockDim.x;
    for (; i < n; i += stride) dst[i] = (bf16)src[i];
}

// ------------------------------------------------------------------- GEMM ---
// C[M,N] = A[M,K] @ Bw[N,K]^T + bias, optional GELU, f32 and/or bf16 outputs.
// Block tile 128x256 (8 waves, 2x4, each wave 64x64 = 16 WMMAs per K-step).
// Double-buffered LDS staged via GLOBAL_LOAD_ASYNC_TO_LDS_B128 (ASYNCcnt).
// M multiple of 128, N multiple of 256, K multiple of 32.
#define BM 128
#define BN 256
#define BK 32

__global__ __launch_bounds__(256) void k_gemm_bf16(
    const bf16* __restrict__ A, const bf16* __restrict__ Bw,
    const float* __restrict__ bias, float* __restrict__ Cf,
    bf16* __restrict__ Cb, int M, int N, int K, int act_gelu)
{
    // [2 stages][ A: BM*BK | B: BN*BK ] bf16
    __shared__ __attribute__((aligned(128))) bf16 smem[2 * (BM * BK + BN * BK)];
    const unsigned STAGE_B = (BM * BK + BN * BK) * 2; // bytes per stage (24576)

    int tid  = threadIdx.x;
    int lane = tid & 31;
    int wave = tid >> 5;       // 0..7
    int wm   = wave >> 2;      // 0..1 -> 64-row slab
    int wn   = wave & 3;       // 0..3 -> 64-col slab
    int half = lane >> 4;
    int l15  = lane & 15;

    long m0 = (long)blockIdx.x * BM;
    long n0 = (long)blockIdx.y * BN;

    const bf16* Abase = A + m0 * K;   // block-local base (byte offsets < 2^31)
    const bf16* Bbase = Bw + n0 * K;

    unsigned ldsBase = (unsigned)(size_t)(void*)&smem[0]; // LDS aperture: low 32 bits = DS offset

    v8f acc[4][4];
    for (int i = 0; i < 4; ++i)
        for (int j = 0; j < 4; ++j) acc[i][j] = vzero8();

    // Issue one stage of async global->LDS copies (6 instructions per lane/wave).
    auto issue_stage = [&](int buf, int k0) {
        unsigned aL = ldsBase + buf * STAGE_B;
        unsigned bL = aL + BM * BK * 2;
        // A tile: 128 rows x 64B = 512 chunks of 16B
        for (int p = 0; p < 2; ++p) {
            int ci = tid + p * 256;
            int row = ci >> 2, cir = ci & 3;
            unsigned go = (unsigned)(row * K + k0) * 2u + (unsigned)cir * 16u;
            unsigned lo = aL + (unsigned)row * 64u + (unsigned)cir * 16u;
            asm volatile("global_load_async_to_lds_b128 %0, %1, %2"
                         :: "v"(lo), "v"(go), "s"(Abase) : "memory");
        }
        // B tile: 256 rows x 64B = 1024 chunks of 16B
        for (int p = 0; p < 4; ++p) {
            int ci = tid + p * 256;
            int row = ci >> 2, cir = ci & 3;
            unsigned go = (unsigned)(row * K + k0) * 2u + (unsigned)cir * 16u;
            unsigned lo = bL + (unsigned)row * 64u + (unsigned)cir * 16u;
            asm volatile("global_load_async_to_lds_b128 %0, %1, %2"
                         :: "v"(lo), "v"(go), "s"(Bbase) : "memory");
        }
    };

    issue_stage(0, 0);
    int nk = K / BK;
    for (int ki = 0; ki < nk; ++ki) {
        int buf = ki & 1;
        if (ki + 1 < nk) {
            issue_stage(buf ^ 1, (ki + 1) * BK);
            // in-order async completion: <=6 outstanding => current stage landed
            asm volatile("s_wait_asynccnt 0x6" ::: "memory");
        } else {
            asm volatile("s_wait_asynccnt 0x0" ::: "memory");
        }
        __syncthreads();

        const bf16* As = &smem[buf * (BM * BK + BN * BK)];
        const bf16* Bs = As + BM * BK;

        AFrag af[4];
        v16bf bfv[4];
        for (int im = 0; im < 4; ++im) {
            int row = wm * 64 + im * 16 + l15;
            af[im].h[0] = *(const v8bf*)&As[row * BK + half * 16];
            af[im].h[1] = *(const v8bf*)&As[row * BK + half * 16 + 8];
        }
        for (int in = 0; in < 4; ++in) {
            int col = wn * 64 + in * 16 + l15;
            bfv[in] = *(const v16bf*)&Bs[col * BK + half * 16];
        }
        for (int im = 0; im < 4; ++im)
            for (int in = 0; in < 4; ++in)
                acc[im][in] = __builtin_amdgcn_wmma_f32_16x16x32_bf16(
                    false, af[im].v, false, bfv[in], (short)0, acc[im][in], false, false);
        __syncthreads();
    }

    for (int in = 0; in < 4; ++in) {
        long col = n0 + wn * 64 + in * 16 + l15;
        float bv = bias ? bias[col] : 0.0f;
        for (int im = 0; im < 4; ++im) {
            for (int r = 0; r < 8; ++r) {
                long row = m0 + wm * 64 + im * 16 + half * 8 + r;
                float v = acc[im][in][r] + bv;
                if (act_gelu) v = gelu_f(v);
                if (Cf) Cf[row * N + col] = v;
                if (Cb) Cb[row * N + col] = (bf16)v;
            }
        }
    }
}

// -------------------------------------------------------------- QKV split ---
__global__ __launch_bounds__(256) void k_qkv_split(const float* __restrict__ qkv,
                                                   bf16* __restrict__ qb,
                                                   bf16* __restrict__ kb,
                                                   bf16* __restrict__ vb)
{
    const int S_ = 2048, D_ = 1024, H_ = 16, DH_ = 64;
    long idx = (long)blockIdx.x * blockDim.x + threadIdx.x;  // over BS*D
    long total = (long)4 * S_ * D_;
    if (idx >= total) return;
    long t = idx / D_;       // token 0..8191
    int  d = (int)(idx % D_);
    int  hh = d / DH_, dh = d % DH_;
    int  b = (int)(t / S_), sp = (int)(t % S_);
    long dst = (((long)b * H_ + hh) * S_ + sp) * DH_ + dh;
    long src = t * (3L * D_);
    qb[dst] = (bf16)qkv[src + d];
    kb[dst] = (bf16)qkv[src + D_ + d];
    vb[dst] = (bf16)qkv[src + 2 * D_ + d];
}

// -------------------------------------------------------------- attention ---
// One wave per (b, h, 16-row q tile). Flash-style online softmax over 32-key
// chunks. q/k/v laid out [B,H,S,64] bf16. Output merged heads [B,S,D] bf16.
__global__ __launch_bounds__(32) void k_attention(
    const bf16* __restrict__ qb, const bf16* __restrict__ kb,
    const bf16* __restrict__ vb, const unsigned char* __restrict__ pad,
    bf16* __restrict__ attnb)
{
    const int S_ = 2048, DH_ = 64, H_ = 16, D_ = 1024;
    const float NEG_INF = -__builtin_inff();

    int lane = threadIdx.x;
    int half = lane >> 4, l15 = lane & 15;
    int qt = blockIdx.x;   // 0..127
    int h  = blockIdx.y;   // 0..15
    int b  = blockIdx.z;   // 0..3
    long bh = ((long)b * H_ + h) * S_;
    int i0 = qt * 16;

    __shared__ __attribute__((aligned(64))) float sc[16 * 32];
    __shared__ __attribute__((aligned(64))) bf16  P[16 * 32];
    __shared__ __attribute__((aligned(64))) bf16  Vt[32 * 64];
    __shared__ float fac[16];
    __shared__ float lsum[16];

    // Q fragments for this q-tile (A layout: lane = row l15, K chunks by half)
    AFrag qf[2];
    {
        long qbase = (bh + i0 + l15) * DH_;
        for (int f = 0; f < 2; ++f) {
            qf[f].h[0] = *(const v8bf*)&qb[qbase + f * 32 + half * 16];
            qf[f].h[1] = *(const v8bf*)&qb[qbase + f * 32 + half * 16 + 8];
        }
    }

    v8f O[4];
    for (int t = 0; t < 4; ++t) O[t] = vzero8();
    float m_run = NEG_INF, l_run = 0.0f;
    int row = l15;
    int qrow_glob = i0 + row;

    int jmax = i0 + 15;
    for (int j0 = 0; j0 <= jmax; j0 += 32) {
        // ---- scores: two 16-key subtiles, each 2 WMMAs over DH=64 ----
        for (int sj = 0; sj < 2; ++sj) {
            v8f s = vzero8();
            int j = j0 + sj * 16 + l15;
            int jc = j < S_ ? j : S_ - 1;
            long kbase = (bh + jc) * DH_;
            for (int f = 0; f < 2; ++f) {
                v16bf kf = *(const v16bf*)&kb[kbase + f * 32 + half * 16];
                s = __builtin_amdgcn_wmma_f32_16x16x32_bf16(
                        false, qf[f].v, false, kf, (short)0, s, false, false);
            }
            for (int r = 0; r < 8; ++r)
                sc[(half * 8 + r) * 32 + sj * 16 + l15] = s[r];
        }
        // ---- stage V chunk (32 keys x 64) into LDS ----
        {
            int j = j0 + lane;
            int jc = j < S_ ? j : S_ - 1;
            long vbase = (bh + jc) * DH_;
            for (int c = 0; c < 4; ++c)
                *(v16bf*)&Vt[lane * DH_ + c * 16] = *(const v16bf*)&vb[vbase + c * 16];
        }
        __syncthreads();

        // ---- online softmax: lane handles row=l15, cols half*16..+15 ----
        float vals[16];
        float pm = NEG_INF;
        for (int c = 0; c < 16; ++c) {
            int j = j0 + half * 16 + c;
            float v = sc[row * 32 + half * 16 + c] * 0.125f;
            int jc = j < S_ ? j : S_ - 1;
            bool mask = (j > qrow_glob) || (j >= S_) || (pad[(long)b * S_ + jc] != 0);
            v = mask ? NEG_INF : v;
            vals[c] = v;
            pm = fmaxf(pm, v);
        }
        float mchunk = fmaxf(pm, __shfl_xor(pm, 16, 32));
        float mnew = fmaxf(m_run, mchunk);
        float factor = (m_run == NEG_INF) ? 0.0f : __expf(m_run - mnew);
        float ps = 0.0f;
        for (int c = 0; c < 16; ++c) {
            float p = (vals[c] == NEG_INF || mnew == NEG_INF)
                          ? 0.0f : __expf(vals[c] - mnew);
            ps += p;
            P[row * 32 + half * 16 + c] = (bf16)p;
        }
        float lchunk = ps + __shfl_xor(ps, 16, 32);
        l_run = l_run * factor + lchunk;
        m_run = mnew;
        if (lane < 16) fac[lane] = factor;
        __syncthreads();

        // ---- rescale accumulators, then P @ V ----
        float fr[8];
        for (int r = 0; r < 8; ++r) fr[r] = fac[half * 8 + r];
        for (int t = 0; t < 4; ++t)
            for (int r = 0; r < 8; ++r) O[t][r] *= fr[r];

        AFrag pf;
        pf.h[0] = *(const v8bf*)&P[l15 * 32 + half * 16];
        pf.h[1] = *(const v8bf*)&P[l15 * 32 + half * 16 + 8];
        for (int t = 0; t < 4; ++t) {
            v16bf vf;
            int dh = t * 16 + l15;
            for (int kk = 0; kk < 16; ++kk)
                vf[kk] = Vt[(half * 16 + kk) * DH_ + dh];
            O[t] = __builtin_amdgcn_wmma_f32_16x16x32_bf16(
                       false, pf.v, false, vf, (short)0, O[t], false, false);
        }
        __syncthreads();
    }

    if (lane < 16) lsum[lane] = l_run;
    __syncthreads();
    for (int t = 0; t < 4; ++t) {
        for (int r = 0; r < 8; ++r) {
            int m = half * 8 + r;
            float l = lsum[m];
            float inv = (l > 0.0f) ? 1.0f / l : 0.0f;
            long i = i0 + m;
            long col = (long)h * DH_ + t * 16 + l15;
            attnb[((long)b * S_ + i) * D_ + col] = (bf16)(O[t][r] * inv);
        }
    }
}

// ----------------------------------------------------- residual + LayerNorm -
__global__ __launch_bounds__(256) void k_add_ln(
    const float* __restrict__ X, const float* __restrict__ R,
    const float* __restrict__ g, const float* __restrict__ bta,
    float* __restrict__ outf, bf16* __restrict__ outb, int Dcols)
{
    __shared__ float red[256];
    long rowbase = (long)blockIdx.x * Dcols;
    int tid = threadIdx.x;

    float s = 0.0f, s2 = 0.0f;
    for (int c = tid; c < Dcols; c += 256) {
        float v = X[rowbase + c] + R[rowbase + c];
        s += v; s2 += v * v;
    }
    red[tid] = s; __syncthreads();
    for (int o = 128; o > 0; o >>= 1) { if (tid < o) red[tid] += red[tid + o]; __syncthreads(); }
    float mean = red[0] / Dcols;
    __syncthreads();
    red[tid] = s2; __syncthreads();
    for (int o = 128; o > 0; o >>= 1) { if (tid < o) red[tid] += red[tid + o]; __syncthreads(); }
    float var = red[0] / Dcols - mean * mean;
    float rstd = rsqrtf(var + 1e-5f);

    for (int c = tid; c < Dcols; c += 256) {
        float v = X[rowbase + c] + R[rowbase + c];
        float y = g[c] * (v - mean) * rstd + bta[c];
        if (outf) outf[rowbase + c] = y;
        if (outb) outb[rowbase + c] = (bf16)y;
    }
}

// ------------------------------------------------------------------ launch ---
extern "C" void kernel_launch(void* const* d_in, const int* in_sizes, int n_in,
                              void* d_out, int out_size, void* d_ws, size_t ws_size,
                              hipStream_t stream) {
    (void)in_sizes; (void)n_in; (void)out_size; (void)ws_size;
    const long Bn = 4, S_ = 2048, D_ = 1024;
    const long BS = Bn * S_;

    const float* x        = (const float*)d_in[0];
    const unsigned char* pad = (const unsigned char*)d_in[1];
    const float* w_qkv    = (const float*)d_in[2];
    const float* b_qkv    = (const float*)d_in[3];
    const float* w_out    = (const float*)d_in[4];
    const float* b_out    = (const float*)d_in[5];
    const float* ln1_g    = (const float*)d_in[6];
    const float* ln1_b    = (const float*)d_in[7];
    const float* w1       = (const float*)d_in[8];
    const float* b1       = (const float*)d_in[9];
    const float* w2       = (const float*)d_in[10];
    const float* b2       = (const float*)d_in[11];
    const float* ln2_g    = (const float*)d_in[12];
    const float* ln2_b    = (const float*)d_in[13];
    float* out            = (float*)d_out;

    size_t off = 0;
    auto alloc = [&](size_t bytes) -> void* {
        void* p = (char*)d_ws + off;
        off += (bytes + 255) & ~(size_t)255;
        return p;
    };

    bf16* xb     = (bf16*)alloc(BS * D_ * 2);
    bf16* wqkvb  = (bf16*)alloc(3072L * 1024 * 2);
    bf16* woutb  = (bf16*)alloc(1024L * 1024 * 2);
    bf16* w1b    = (bf16*)alloc(4096L * 1024 * 2);
    bf16* w2b    = (bf16*)alloc(1024L * 4096 * 2);
    float* qkvf  = (float*)alloc(BS * 3072 * 4);   // reused as FFN hidden (bf16)
    bf16* hbuf   = (bf16*)qkvf;                    // BS*4096*2 <= BS*3072*4
    bf16* qb     = (bf16*)alloc(BS * D_ * 2);
    bf16* kb     = (bf16*)alloc(BS * D_ * 2);
    bf16* vb     = (bf16*)alloc(BS * D_ * 2);
    bf16* attnb  = (bf16*)alloc(BS * D_ * 2);
    float* abuf  = (float*)alloc(BS * D_ * 4);     // attn out; reused as FFN out
    float* fbuf  = abuf;
    float* x1f   = (float*)alloc(BS * D_ * 4);
    bf16* x1b    = (bf16*)alloc(BS * D_ * 2);

    // ---- convert inputs/weights to bf16 ----
    {
        long n = BS * D_;
        k_f32_to_bf16<<<(int)((n + 255) / 256), 256, 0, stream>>>(x, xb, n);
        n = 3072L * 1024;
        k_f32_to_bf16<<<(int)((n + 255) / 256), 256, 0, stream>>>(w_qkv, wqkvb, n);
        n = 1024L * 1024;
        k_f32_to_bf16<<<(int)((n + 255) / 256), 256, 0, stream>>>(w_out, woutb, n);
        n = 4096L * 1024;
        k_f32_to_bf16<<<(int)((n + 255) / 256), 256, 0, stream>>>(w1, w1b, n);
        n = 1024L * 4096;
        k_f32_to_bf16<<<(int)((n + 255) / 256), 256, 0, stream>>>(w2, w2b, n);
    }

    // ---- qkv = x @ w_qkv^T + b_qkv ----
    k_gemm_bf16<<<dim3(BS / BM, 3072 / BN), 256, 0, stream>>>(
        xb, wqkvb, b_qkv, qkvf, nullptr, (int)BS, 3072, 1024, 0);

    // ---- split into per-head q/k/v bf16 ----
    {
        long n = BS * D_;
        k_qkv_split<<<(int)((n + 255) / 256), 256, 0, stream>>>(qkvf, qb, kb, vb);
    }

    // ---- attention ----
    k_attention<<<dim3(S_ / 16, 16, Bn), 32, 0, stream>>>(qb, kb, vb, pad, attnb);

    // ---- a = attn @ w_out^T + b_out ----
    k_gemm_bf16<<<dim3(BS / BM, 1024 / BN), 256, 0, stream>>>(
        attnb, woutb, b_out, abuf, nullptr, (int)BS, 1024, 1024, 0);

    // ---- x1 = LN(x + a) ----
    k_add_ln<<<(int)BS, 256, 0, stream>>>(x, abuf, ln1_g, ln1_b, x1f, x1b, 1024);

    // ---- h = gelu(x1 @ w1^T + b1) (bf16) ----
    k_gemm_bf16<<<dim3(BS / BM, 4096 / BN), 256, 0, stream>>>(
        x1b, w1b, b1, nullptr, hbuf, (int)BS, 4096, 1024, 1);

    // ---- f = h @ w2^T + b2 ----
    k_gemm_bf16<<<dim3(BS / BM, 1024 / BN), 256, 0, stream>>>(
        hbuf, w2b, b2, fbuf, nullptr, (int)BS, 1024, 4096, 0);

    // ---- out = LN(x1 + f) ----
    k_add_ln<<<(int)BS, 256, 0, stream>>>(x1f, fbuf, ln2_g, ln2_b, out, nullptr, 1024);
}
